// GraphTransformer_27084063769019
// MI455X (gfx1250) — matrix operand order repaired and checked by reading, not compiled, not run on hardware
//
#include <hip/hip_runtime.h>
#include <math.h>

typedef __attribute__((ext_vector_type(16))) _Float16 v16h;
typedef __attribute__((ext_vector_type(8)))  _Float16 v8h;
typedef __attribute__((ext_vector_type(8)))  float    v8f;

#define NN 50000
#define NE 800000
#define FD 64          // feature width per layer (H*D = 64 for both layers)
#define MT (NN/16)     // 3125 row tiles, exact

// ---------------- elementwise helpers ----------------
__global__ void k_f32_to_f16(const float* __restrict__ in, _Float16* __restrict__ out, int n) {
    int i = blockIdx.x * blockDim.x + threadIdx.x;
    if (i < n) out[i] = (_Float16)in[i];
}

__global__ void k_relu_to_f16(const float* __restrict__ in, _Float16* __restrict__ out, int n) {
    int i = blockIdx.x * blockDim.x + threadIdx.x;
    if (i < n) { float v = in[i]; out[i] = (_Float16)(v > 0.f ? v : 0.f); }
}

__global__ void k_fill_u32(unsigned* __restrict__ p, unsigned v, int n) {
    int i = blockIdx.x * blockDim.x + threadIdx.x;
    if (i < n) p[i] = v;
}

__global__ void k_fill_f32(float* __restrict__ p, float v, int n) {
    int i = blockIdx.x * blockDim.x + threadIdx.x;
    if (i < n) p[i] = v;
}

// ---------------- fused 4-weight WMMA GEMM, LDS-staged B ----------------
// Yw[M,64] = Xh[M,K] * Ww[64,K]^T + bw  for w = 0..3.
// 8 waves per block share one LDS copy of all 4 weight matrices, stored
// FRAGMENT-MAJOR: sB[tile][kstep][lane][16 halves]. Each lane's B fragment is
// a contiguous 32B chunk; a wave's 32 fragments are a contiguous 1KB region
// -> conflict-free, fully coalesced ds_load_b128 pairs. A fragments for the
// whole K extent are preloaded once per wave (X read exactly once).
// Fragment layouts per CDNA5 ISA 7.12.2 (wave32):
//   A 16x32 f16 : lane holds row (lane&15); lanes<16 K {0..7,16..23},
//                 lanes>=16 K {8..15,24..31}  -> two v8h loads
//   B 32x16 f16 : lane holds column (lane&15); lanes<16 K {0..7,16..23},
//                 lanes>=16 K {8..15,24..31}
//   C/D 16x16 f32: VGPR r -> row r+8*(lane>>4), col (lane&15)
template <int K>
__global__ __launch_bounds__(256)
void k_gemm4_wmma(const _Float16* __restrict__ X,
                  const _Float16* __restrict__ W,   // [4][64][K] contiguous
                  const float* __restrict__ b0, const float* __restrict__ b1,
                  const float* __restrict__ b2, const float* __restrict__ b3,
                  float* __restrict__ Y0, float* __restrict__ Y1,
                  float* __restrict__ Y2, float* __restrict__ Y3,
                  int Mtiles) {
    constexpr int KS     = K / 32;        // k-steps (4 for K=128, 2 for K=64)
    constexpr int CHUNKS = 16 * KS * 32;  // 32B fragments: 16 tiles x KS x 32 lanes
    __shared__ _Float16 sB[CHUNKS * 16];  // 64KB (K=128) / 32KB (K=64)

    // ---- cooperative staging: global W -> fragment-major LDS ----
    for (int c = threadIdx.x; c < CHUNKS; c += 256) {
        const int l  = c & 31;            // lane the fragment belongs to
        const int ks = (c >> 5) % KS;     // k-step
        const int t  = c / (32 * KS);     // tile: t = w*4 + nt
        const _Float16* src = W + (size_t)(t * 16 + (l & 15)) * K + ks * 32 + (l >> 4) * 8;
        uint4 d0 = *(const uint4*)src;        // 16B
        uint4 d1 = *(const uint4*)(src + 16); // 16B
        *(uint4*)(sB + (size_t)c * 16)     = d0;
        *(uint4*)(sB + (size_t)c * 16 + 8) = d1;
    }
    __syncthreads();

    const int lane = threadIdx.x & 31;
    const int wv   = threadIdx.x >> 5;
    const int mt   = blockIdx.x * 8 + wv;
    if (mt >= Mtiles) return;
    const int l15 = lane & 15;
    const int hi  = lane >> 4;            // 0 or 1: half-lane group

    // ---- preload A fragments for all k-steps (read X exactly once) ----
    const _Float16* xrow = X + (size_t)(mt * 16 + l15) * K + hi * 8;
    v16h a[KS];
    #pragma unroll
    for (int ks = 0; ks < KS; ++ks) {
        v8h alo = *(const v8h*)(xrow + ks * 32);
        v8h ahi = *(const v8h*)(xrow + ks * 32 + 16);
        #pragma unroll
        for (int i = 0; i < 8; ++i) { a[ks][i] = alo[i]; a[ks][8 + i] = ahi[i]; }
    }

    const float* bias[4] = { b0, b1, b2, b3 };
    float*       Ys[4]   = { Y0, Y1, Y2, Y3 };

    #pragma unroll
    for (int w = 0; w < 4; ++w) {
        #pragma unroll
        for (int nt = 0; nt < 4; ++nt) {
            const int t = w * 4 + nt;
            v8f acc = {};
            #pragma unroll
            for (int ks = 0; ks < KS; ++ks) {
                v16h b = *(const v16h*)(sB + (size_t)((t * KS + ks) * 32 + lane) * 16);
                acc = __builtin_amdgcn_wmma_f32_16x16x32_f16(
                    false, a[ks], false, b, (short)0, acc, false, false);
            }
            float* yb = Ys[w] + (size_t)(mt * 16 + hi * 8) * FD + nt * 16 + l15;
            float bv = bias[w][nt * 16 + l15];
            #pragma unroll
            for (int r = 0; r < 8; ++r)
                yb[(size_t)r * FD] = acc[r] + bv;
        }
    }
}

// ---------------- order-preserving float<->uint for atomic max ----------------
__device__ __forceinline__ unsigned enc_f32(float f) {
    unsigned u = __float_as_uint(f);
    return (u & 0x80000000u) ? ~u : (u | 0x80000000u);
}
__device__ __forceinline__ float dec_f32(unsigned k) {
    return (k & 0x80000000u) ? __uint_as_float(k & 0x7fffffffu) : __uint_as_float(~k);
}
#define ENC_NEG_INF 0x007FFFFFu   // enc(-inf)

// ---------------- edge-wise attention kernels ----------------
// alpha[e,h] = dot(q[dst[e],h,:], k[src[e],h,:]) * scale
__global__ void k_edge_dot(const float* __restrict__ q, const float* __restrict__ kk,
                           const int* __restrict__ src, const int* __restrict__ dst,
                           float* __restrict__ alpha, int H, int D, float scale, int n) {
    int idx = blockIdx.x * blockDim.x + threadIdx.x;
    if (idx >= n) return;
    int e = idx / H, h = idx - e * H;
    const float4* qv = (const float4*)(q  + (size_t)dst[e] * FD + h * D);
    const float4* kv = (const float4*)(kk + (size_t)src[e] * FD + h * D);
    float s = 0.f;
    for (int i = 0; i < (D >> 2); ++i) {
        float4 a = qv[i], b = kv[i];
        s += a.x * b.x + a.y * b.y + a.z * b.z + a.w * b.w;
    }
    alpha[idx] = s * scale;
}

__global__ void k_seg_max(const float* __restrict__ alpha, const int* __restrict__ dst,
                          unsigned* __restrict__ amax, int H, int n) {
    int idx = blockIdx.x * blockDim.x + threadIdx.x;
    if (idx >= n) return;
    int e = idx / H, h = idx - e * H;
    atomicMax(&amax[(size_t)dst[e] * H + h], enc_f32(alpha[idx]));
}

__global__ void k_exp_denom(float* __restrict__ alpha, const int* __restrict__ dst,
                            const unsigned* __restrict__ amax, float* __restrict__ denom,
                            int H, int n) {
    int idx = blockIdx.x * blockDim.x + threadIdx.x;
    if (idx >= n) return;
    int e = idx / H, h = idx - e * H;
    float m = dec_f32(amax[(size_t)dst[e] * H + h]);
    float ex = __expf(alpha[idx] - m);
    alpha[idx] = ex;                       // in-place: alpha now holds exp()
    atomicAdd(&denom[(size_t)dst[e] * H + h], ex);
}

__global__ void k_norm(float* __restrict__ alpha, const int* __restrict__ dst,
                       const float* __restrict__ denom, int H, int n) {
    int idx = blockIdx.x * blockDim.x + threadIdx.x;
    if (idx >= n) return;
    int e = idx / H, h = idx - e * H;
    alpha[idx] = alpha[idx] / denom[(size_t)dst[e] * H + h];
}

// out[dst,h,:] += wgt[e,h] * v[src,h,:]  (16 threads per edge, 4 floats each)
__global__ void k_aggregate(const float* __restrict__ wgt, const float* __restrict__ v,
                            const int* __restrict__ src, const int* __restrict__ dst,
                            float* __restrict__ out, int H, int D, int n) {
    int idx = blockIdx.x * blockDim.x + threadIdx.x;
    if (idx >= n) return;
    int e = idx >> 4, j = idx & 15;        // j indexes a 4-float chunk of the 64-dim value
    int h = (j * 4) / D;
    float w = wgt[(size_t)e * H + h];
    float4 vv = *(const float4*)(v + (size_t)src[e] * FD + j * 4);
    float* op = out + (size_t)dst[e] * FD + j * 4;
    atomicAdd(op + 0, w * vv.x);
    atomicAdd(op + 1, w * vv.y);
    atomicAdd(op + 2, w * vv.z);
    atomicAdd(op + 3, w * vv.w);
}

// ---------------- host-side orchestration ----------------
extern "C" void kernel_launch(void* const* d_in, const int* in_sizes, int n_in,
                              void* d_out, int out_size, void* d_ws, size_t ws_size,
                              hipStream_t stream) {
    const float* x   = (const float*)d_in[0];
    const int*   ei  = (const int*)d_in[1];
    const int*   src = ei;
    const int*   dst = ei + NE;
    const float* Wq1 = (const float*)d_in[2];  const float* bq1 = (const float*)d_in[3];
    const float* Wk1 = (const float*)d_in[4];  const float* bk1 = (const float*)d_in[5];
    const float* Wv1 = (const float*)d_in[6];  const float* bv1 = (const float*)d_in[7];
    const float* Ws1 = (const float*)d_in[8];  const float* bs1 = (const float*)d_in[9];
    const float* Wq2 = (const float*)d_in[10]; const float* bq2 = (const float*)d_in[11];
    const float* Wk2 = (const float*)d_in[12]; const float* bk2 = (const float*)d_in[13];
    const float* Wv2 = (const float*)d_in[14]; const float* bv2 = (const float*)d_in[15];
    const float* Ws2 = (const float*)d_in[16]; const float* bs2 = (const float*)d_in[17];
    float* out = (float*)d_out;

    // carve workspace (aligned 256B)
    char* wsp = (char*)d_ws;
    auto carve = [&](size_t bytes) {
        char* p = wsp; wsp += (bytes + 255) & ~(size_t)255; return (void*)p;
    };
    _Float16* Xh  = (_Float16*)carve((size_t)NN * 128 * 2);   // x in f16
    _Float16* W1h = (_Float16*)carve((size_t)4 * 64 * 128 * 2);
    _Float16* W2h = (_Float16*)carve((size_t)4 * 64 * 64 * 2);
    float*    qb  = (float*)carve((size_t)NN * FD * 4);       // reused both layers
    float*    kb  = (float*)carve((size_t)NN * FD * 4);
    float*    vb  = (float*)carve((size_t)NN * FD * 4);
    float*    h1  = (float*)carve((size_t)NN * FD * 4);       // layer-1 accum (init = skip)
    _Float16* Hh  = (_Float16*)carve((size_t)NN * FD * 2);    // relu(h1) in f16
    float*    al  = (float*)carve((size_t)NE * 4 * 4);        // alpha / exp / weights
    unsigned* amx = (unsigned*)carve((size_t)NN * 4 * 4);
    float*    den = (float*)carve((size_t)NN * 4 * 4);

    const int T = 256;
    auto cdiv = [](int a, int b) { return (a + b - 1) / b; };
    const int gemm_blocks = cdiv(MT, 8);   // 8 waves (M-tiles) per block

    // ---- precision conversion ----
    k_f32_to_f16<<<cdiv(NN * 128, T), T, 0, stream>>>(x, Xh, NN * 128);
    k_f32_to_f16<<<cdiv(4 * 64 * 128, T), T, 0, stream>>>(Wq1, W1h + 0 * 8192, 8192);
    k_f32_to_f16<<<cdiv(4 * 64 * 128, T), T, 0, stream>>>(Wk1, W1h + 1 * 8192, 8192);
    k_f32_to_f16<<<cdiv(4 * 64 * 128, T), T, 0, stream>>>(Wv1, W1h + 2 * 8192, 8192);
    k_f32_to_f16<<<cdiv(4 * 64 * 128, T), T, 0, stream>>>(Ws1, W1h + 3 * 8192, 8192);

    // ---- layer 1: fused q/k/v/skip projections (LDS-staged weights) ----
    k_gemm4_wmma<128><<<gemm_blocks, 256, 0, stream>>>(
        Xh, W1h, bq1, bk1, bv1, bs1, qb, kb, vb, h1, MT);   // skip -> h1

    // ---- layer 1 attention: H=4, D=16, scale = 1/sqrt(16) ----
    {
        const int H = 4, D = 16, nEH = NE * 4;
        k_edge_dot<<<cdiv(nEH, T), T, 0, stream>>>(qb, kb, src, dst, al, H, D, 0.25f, nEH);
        k_fill_u32<<<cdiv(NN * 4, T), T, 0, stream>>>(amx, ENC_NEG_INF, NN * 4);
        k_fill_f32<<<cdiv(NN * 4, T), T, 0, stream>>>(den, 0.f, NN * 4);
        k_seg_max<<<cdiv(nEH, T), T, 0, stream>>>(al, dst, amx, H, nEH);
        k_exp_denom<<<cdiv(nEH, T), T, 0, stream>>>(al, dst, amx, den, H, nEH);
        k_norm<<<cdiv(nEH, T), T, 0, stream>>>(al, dst, den, H, nEH);
        k_aggregate<<<cdiv(NE * 16, T), T, 0, stream>>>(al, vb, src, dst, h1, H, D, NE * 16);
    }

    // ---- relu + f16 for layer 2 ----
    k_relu_to_f16<<<cdiv(NN * FD, T), T, 0, stream>>>(h1, Hh, NN * FD);

    // ---- layer 2: fused q/k/v/skip projections ----
    k_f32_to_f16<<<cdiv(4 * 64 * 64, T), T, 0, stream>>>(Wq2, W2h + 0 * 4096, 4096);
    k_f32_to_f16<<<cdiv(4 * 64 * 64, T), T, 0, stream>>>(Wk2, W2h + 1 * 4096, 4096);
    k_f32_to_f16<<<cdiv(4 * 64 * 64, T), T, 0, stream>>>(Wv2, W2h + 2 * 4096, 4096);
    k_f32_to_f16<<<cdiv(4 * 64 * 64, T), T, 0, stream>>>(Ws2, W2h + 3 * 4096, 4096);
    k_gemm4_wmma<64><<<gemm_blocks, 256, 0, stream>>>(
        Hh, W2h, bq2, bk2, bv2, bs2, qb, kb, vb, out, MT);  // skip -> out

    // ---- layer 2 attention: H=1, D=64, scale = 1/sqrt(64) ----
    {
        const int H = 1, D = 64, nEH = NE;
        k_edge_dot<<<cdiv(nEH, T), T, 0, stream>>>(qb, kb, src, dst, al, H, D, 0.125f, nEH);
        k_fill_u32<<<cdiv(NN, T), T, 0, stream>>>(amx, ENC_NEG_INF, NN);
        k_fill_f32<<<cdiv(NN, T), T, 0, stream>>>(den, 0.f, NN);
        k_seg_max<<<cdiv(nEH, T), T, 0, stream>>>(al, dst, amx, 1, nEH);
        k_exp_denom<<<cdiv(nEH, T), T, 0, stream>>>(al, dst, amx, den, 1, nEH);
        k_norm<<<cdiv(nEH, T), T, 0, stream>>>(al, dst, den, 1, nEH);
        k_aggregate<<<cdiv(NE * 16, T), T, 0, stream>>>(al, vb, src, dst, out, H, D, NE * 16);
    }
    (void)in_sizes; (void)n_in; (void)out_size; (void)ws_size;
}
